// GNN_25211458028088
// MI455X (gfx1250) — matrix operand order repaired and checked by reading, not compiled, not run on hardware
//
#include <hip/hip_runtime.h>

// ---------- types ----------
typedef unsigned short u16;
typedef __attribute__((ext_vector_type(16))) __bf16 v16bf;
typedef __attribute__((ext_vector_type(8)))  float  v8f;
typedef __attribute__((ext_vector_type(16))) u16    v16u;
typedef __attribute__((ext_vector_type(8)))  u16    v8u;

__device__ __forceinline__ u16 f32_to_bf16_rne(float f) {
    unsigned u = __builtin_bit_cast(unsigned, f);
    u += 0x7FFFu + ((u >> 16) & 1u);
    return (u16)(u >> 16);
}
__device__ __forceinline__ float bf16_to_f32(u16 h) {
    unsigned u = ((unsigned)h) << 16;
    return __builtin_bit_cast(float, u);
}

// ---------- cast f32 -> bf16 ----------
__global__ void cast_f32_to_bf16(const float* __restrict__ in, u16* __restrict__ out, long n) {
    long i = (long)blockIdx.x * blockDim.x + threadIdx.x;
    long stride = (long)gridDim.x * blockDim.x;
    for (; i < n; i += stride) out[i] = f32_to_bf16_rne(in[i]);
}

// ---------- WMMA GEMM: C[M,Nc] = A[M,Ktot] * W[Nc,Ktot]^T (+bias,relu,residual) ----------
// Block: 8 waves, each wave owns a 32(M) x 64(N) tile -> block tile 256 x 64.
// The 64-column weight tile (64 x Ktot bf16, <= 66 KB) is staged once into LDS and
// shared by all 8 waves; the K-loop reads B via ds_load (padded stride for bank
// stagger) and A via global_load_b128 with prefetch. 8 independent WMMAs per K-step.
// A is split into HAS_A1 ? 2 : 1 segments of width Kseg (handles [h | Ah] concat).
// FUSE: v = Cf[idx] + relu(acc + bias[col]) written in place (race-free: each lane
// reads only the element it writes). Optional bf16 mirror of the output in Cbf.
template<bool HAS_A1, bool FUSE>
__global__ __launch_bounds__(256)
void wmma_gemm_bf16(const u16* __restrict__ A0, const u16* __restrict__ A1,
                    const u16* __restrict__ W,  const float* __restrict__ bias,
                    float* __restrict__ Cf, u16* __restrict__ Cbf,
                    int M, int Nc, int Kseg)
{
    constexpr int KMAX = 512;
    constexpr int PAD  = 16;                          // halves: +8 dword-banks per column
    __shared__ __align__(32) u16 sB[64 * (KMAX + PAD)];

    const int tid  = threadIdx.x;
    const int lane = tid & 31;
    const int wave = tid >> 5;
    const int Ktot = HAS_A1 ? 2 * Kseg : Kseg;
    const int KP   = Ktot + PAD;
    const int col0 = blockIdx.x * 64;

    // ---- cooperative stage of the 64 x Ktot weight tile into LDS ----
    {
        const int cpk    = Ktot / 8;                  // 16-byte chunks per column
        const int chunks = 64 * cpk;
        for (int c = tid; c < chunks; c += 256) {
            const int col = c / cpk;
            const int kc  = (c - col * cpk) * 8;
            v8u d = *(const v8u*)(W + (size_t)(col0 + col) * Ktot + kc);
            *(v8u*)(sB + col * KP + kc) = d;
        }
    }
    __syncthreads();

    const int row0 = blockIdx.y * 256 + wave * 32;
    if (row0 >= M) return;                            // after barrier; wave-uniform

    // A (16x32 bf16): lanes 0-15 row m K 0..15; lanes 16-31 K 16..31
    const int ak   = (lane >> 4) * 16;
    // B (32x16 bf16): lane = column; lanes 0-15 K 0..15, lanes 16-31 K 16..31
    const int bk   = (lane >> 4) * 16;
    const int bcol = lane & 15;
    const u16* sBl = sB + bcol * KP + bk;             // per-lane LDS base

    v8f zero = {};
    v8f acc[2][4];
#pragma unroll
    for (int m = 0; m < 2; ++m)
#pragma unroll
        for (int j = 0; j < 4; ++j) acc[m][j] = zero;

    const u16* Abase[2][2];                           // [seg][m]
#pragma unroll
    for (int m = 0; m < 2; ++m) {
        const int arow = row0 + (lane & 15) + m * 16;
        Abase[0][m] = A0 + (size_t)arow * Kseg;
        Abase[1][m] = (HAS_A1 ? A1 : A0) + (size_t)arow * Kseg;
    }

    const int nseg = HAS_A1 ? 2 : 1;
    for (int seg = 0; seg < nseg; ++seg) {
        const int koff = seg * Kseg;                  // column offset into W / sB
        for (int kk = 0; kk < Kseg; kk += 32) {
            // batch all loads for this K-step first (max memory-level parallelism)
            v16bf a[2];
#pragma unroll
            for (int m = 0; m < 2; ++m) {
                union { v8u h[2]; v16u full; } ua;
                ua.h[0] = *(const v8u*)(Abase[seg][m] + kk + ak);
                ua.h[1] = *(const v8u*)(Abase[seg][m] + kk + ak + 8);
                a[m] = __builtin_bit_cast(v16bf, ua.full);
            }
            if (kk + 32 < Kseg) {
                __builtin_prefetch(Abase[seg][0] + kk + 32 + ak, 0, 1);
                __builtin_prefetch(Abase[seg][1] + kk + 32 + ak, 0, 1);
            }
            v16bf b[4];
#pragma unroll
            for (int j = 0; j < 4; ++j)
                b[j] = __builtin_bit_cast(v16bf,
                        *(const v16u*)(sBl + (size_t)j * 16 * KP + koff + kk));
            // 8 independent WMMAs
#pragma unroll
            for (int j = 0; j < 4; ++j)
#pragma unroll
                for (int m = 0; m < 2; ++m)
                    acc[m][j] = __builtin_amdgcn_wmma_f32_16x16x32_bf16(
                        false, a[m], false, b[j], (short)0, acc[m][j], false, false);
        }
    }

    // Epilogue. C/D layout: VGPR r, lane l -> M = r + 8*(l>>4), N = l&15.
#pragma unroll
    for (int m = 0; m < 2; ++m) {
        const int rbase = row0 + m * 16 + ((lane >> 4) << 3);
#pragma unroll
        for (int j = 0; j < 4; ++j) {
            const int col = col0 + j * 16 + (lane & 15);
            const float bv = FUSE ? bias[col] : 0.0f;
#pragma unroll
            for (int r = 0; r < 8; ++r) {
                const int row = rbase + r;
                if (row < M) {
                    const size_t idx = (size_t)row * Nc + col;
                    float v = acc[m][j][r];
                    if (FUSE) {
                        v += bv;
                        v = v > 0.0f ? v : 0.0f;
                        v += Cf[idx];
                    }
                    Cf[idx] = v;
                    if (Cbf) Cbf[idx] = f32_to_bf16_rne(v);
                }
            }
        }
    }
}

// ---------- edge scatter: agg[dst] += h[src] * w  (one 256-thread block per edge) ----------
__global__ __launch_bounds__(256)
void adj_scatter(const u16* __restrict__ hbf, const int* __restrict__ src,
                 const int* __restrict__ dst, const float* __restrict__ ew,
                 float* __restrict__ agg)
{
    const int e = blockIdx.x;
    const int f = threadIdx.x;          // 256 features
    const int s = src[e];
    const int d = dst[e];
    const float w = ew[e];
    const float v = bf16_to_f32(hbf[(size_t)s * 256 + f]) * w;
    atomicAdd(&agg[(size_t)d * 256 + f], v);
}

// ---------- row-wise L2 normalize, 128 cols, one wave32 per row ----------
__global__ __launch_bounds__(256)
void l2norm_rows(const float* __restrict__ in, float* __restrict__ out, int M)
{
    const int wave = threadIdx.x >> 5;
    const int lane = threadIdx.x & 31;
    const int row  = blockIdx.x * 8 + wave;
    if (row >= M) return;
    const float4 v = ((const float4*)(in + (size_t)row * 128))[lane];
    float s = v.x * v.x + v.y * v.y + v.z * v.z + v.w * v.w;
#pragma unroll
    for (int m = 16; m; m >>= 1) s += __shfl_xor(s, m, 32);
    const float r = rsqrtf(s);
    float4 o;
    o.x = v.x * r; o.y = v.y * r; o.z = v.z * r; o.w = v.w * r;
    ((float4*)(out + (size_t)row * 128))[lane] = o;
}

// ---------- host driver ----------
extern "C" void kernel_launch(void* const* d_in, const int* in_sizes, int n_in,
                              void* d_out, int out_size, void* d_ws, size_t ws_size,
                              hipStream_t stream) {
    const float* x    = (const float*)d_in[0];
    const int*   esrc = (const int*)d_in[1];
    const int*   edst = (const int*)d_in[2];
    const float* ew   = (const float*)d_in[3];
    const float* Wemb = (const float*)d_in[4];
    const float* Wg1  = (const float*)d_in[5];
    const float* bg1  = (const float*)d_in[6];
    const float* Wg2  = (const float*)d_in[7];
    const float* bg2  = (const float*)d_in[8];
    const float* Wfc  = (const float*)d_in[9];

    const int IN_FEAT = 256, HID = 256, OUT_FEAT = 128;
    const int N = in_sizes[0] / IN_FEAT;
    const int E = in_sizes[1];

    // workspace carve (256B aligned)
    char* p = (char*)d_ws;
    auto carve = [&](size_t bytes) -> char* {
        char* r = p;
        p += (bytes + 255) & ~(size_t)255;
        return r;
    };
    u16*   x_bf    = (u16*)  carve((size_t)N * IN_FEAT * 2);
    u16*   Wemb_bf = (u16*)  carve((size_t)HID * IN_FEAT * 2);
    u16*   Wg1_bf  = (u16*)  carve((size_t)HID * 2 * HID * 2);
    u16*   Wg2_bf  = (u16*)  carve((size_t)HID * 2 * HID * 2);
    u16*   Wfc_bf  = (u16*)  carve((size_t)OUT_FEAT * HID * 2);
    float* h       = (float*)carve((size_t)N * HID * 4);
    u16*   hbf_a   = (u16*)  carve((size_t)N * HID * 2);
    u16*   hbf_b   = (u16*)  carve((size_t)N * HID * 2);
    float* agg     = (float*)carve((size_t)N * HID * 4);
    u16*   agg_bf  = (u16*)  carve((size_t)N * HID * 2);
    float* outr    = agg;    // agg's last read is the cast before gc2; safe reuse for fc output

    auto cast = [&](const float* srcp, u16* dstp, long n) {
        int blocks = (int)((n + 255) / 256);
        if (blocks > 2048) blocks = 2048;
        cast_f32_to_bf16<<<blocks, 256, 0, stream>>>(srcp, dstp, n);
    };

    cast(x,    x_bf,    (long)N * IN_FEAT);
    cast(Wemb, Wemb_bf, (long)HID * IN_FEAT);
    cast(Wg1,  Wg1_bf,  (long)HID * 2 * HID);
    cast(Wg2,  Wg2_bf,  (long)HID * 2 * HID);
    cast(Wfc,  Wfc_bf,  (long)OUT_FEAT * HID);

    const dim3 blk(256);
    const dim3 g_h(HID / 64, (N + 255) / 256);        // 256-wide outputs
    const dim3 g_o(OUT_FEAT / 64, (N + 255) / 256);   // 128-wide output

    // h = x @ Wemb^T   (f32 + bf16 copies)
    wmma_gemm_bf16<false, false><<<g_h, blk, 0, stream>>>(
        x_bf, nullptr, Wemb_bf, nullptr, h, hbf_a, N, HID, IN_FEAT);

    // layer 1: agg = A@h ; h += relu([h|agg] @ Wg1^T + b1)
    hipMemsetAsync(agg, 0, (size_t)N * HID * 4, stream);
    adj_scatter<<<E, 256, 0, stream>>>(hbf_a, esrc, edst, ew, agg);
    cast(agg, agg_bf, (long)N * HID);
    wmma_gemm_bf16<true, true><<<g_h, blk, 0, stream>>>(
        hbf_a, agg_bf, Wg1_bf, bg1, h, hbf_b, N, HID, HID);

    // layer 2
    hipMemsetAsync(agg, 0, (size_t)N * HID * 4, stream);
    adj_scatter<<<E, 256, 0, stream>>>(hbf_b, esrc, edst, ew, agg);
    cast(agg, agg_bf, (long)N * HID);
    wmma_gemm_bf16<true, true><<<g_h, blk, 0, stream>>>(
        hbf_b, agg_bf, Wg2_bf, bg2, h, hbf_a, N, HID, HID);

    // out = h @ Wfc^T, then row-L2 normalize into d_out
    wmma_gemm_bf16<false, false><<<g_o, blk, 0, stream>>>(
        hbf_a, nullptr, Wfc_bf, nullptr, outr, nullptr, N, OUT_FEAT, HID);
    l2norm_rows<<<(N + 7) / 8, 256, 0, stream>>>(outr, (float*)d_out, N);
}
